// BlockSeries_58205396795417
// MI455X (gfx1250) — compile-verified, matrix operand order checked by reading, not run on hardware
//
#include <hip/hip_runtime.h>
#include <hip/hip_bf16.h>
#include <stdint.h>
#include <stddef.h>

// ---------------------------------------------------------------------------
// Problem constants (match reference)
// ---------------------------------------------------------------------------
#define N_VOX   500000
#define C_IN    32
#define C_OUT4  64
#define NSLOT   256        // slots per channel for partial-stat atomics
#define BN_EPS  1e-4f

typedef __attribute__((ext_vector_type(16))) __bf16 v16bf;
typedef __attribute__((ext_vector_type(8)))  float  v8f;

union ABReg { v16bf v; uint4 q[2]; };
union BF8  { uint4 q; __bf16 h[8]; };

static __device__ __forceinline__ v8f wmma_bf16(const ABReg& a, const ABReg& b,
                                                v8f c) {
  return __builtin_amdgcn_wmma_f32_16x16x32_bf16(
      false, a.v, false, b.v, (short)0, c, false, false);
}

// ---------------------------------------------------------------------------
// Utility kernels
// ---------------------------------------------------------------------------
__global__ void zero_f32_kernel(float* __restrict__ p, long n) {
  long i = (long)blockIdx.x * blockDim.x + threadIdx.x;
  long stride = (long)gridDim.x * blockDim.x;
  for (; i < n; i += stride) p[i] = 0.0f;
}

// feats f32 -> padded bf16 activations, 8 elements/thread; row N_VOX zeroed.
__global__ void cast_feats_kernel(const float* __restrict__ f,
                                  __bf16* __restrict__ xa) {
  const int total8 = (N_VOX + 1) * C_IN / 8;   // divisible exactly
  uint4* out = (uint4*)xa;
  for (int i = blockIdx.x * blockDim.x + threadIdx.x; i < total8;
       i += gridDim.x * blockDim.x) {
    BF8 o;
    if (i < N_VOX * C_IN / 8) {
      const float4 f0 = ((const float4*)f)[2 * i];
      const float4 f1 = ((const float4*)f)[2 * i + 1];
      o.h[0] = (__bf16)f0.x; o.h[1] = (__bf16)f0.y;
      o.h[2] = (__bf16)f0.z; o.h[3] = (__bf16)f0.w;
      o.h[4] = (__bf16)f1.x; o.h[5] = (__bf16)f1.y;
      o.h[6] = (__bf16)f1.z; o.h[7] = (__bf16)f1.w;
    } else {
      o.q = make_uint4(0u, 0u, 0u, 0u);
    }
    out[i] = o.q;
  }
}

// W (27, 32, COUT) f32 -> Wt (27, COUT, 32) bf16  (transposed for B-tile loads)
__global__ void prep_wt_kernel(const float* __restrict__ W,
                               __bf16* __restrict__ wt, int cout) {
  const int total = 27 * cout * 32;
  for (int i = blockIdx.x * blockDim.x + threadIdx.x; i < total;
       i += gridDim.x * blockDim.x) {
    int k   = i / (cout * 32);
    int rem = i - k * cout * 32;
    int n   = rem >> 5;
    int c   = rem & 31;
    wt[i] = (__bf16)W[(k * 32 + c) * cout + n];
  }
}

// ---------------------------------------------------------------------------
// Sparse submanifold conv: rulebook gather + bf16 WMMA.
// One wave owns a 2M x 2N register tile (32 voxels x 32 channels):
//   4 accumulator chains, A/B each reused twice -> 2 vmem loads per WMMA.
// Block = 32 voxels; COUT/32 waves per block (disjoint channel pairs).
// Fused per-channel sum/sumsq partials via slotted global atomics.
// ---------------------------------------------------------------------------
template <int COUT>
__global__ __launch_bounds__(32 * (COUT / 32))
void conv_wmma_kernel(const __bf16* __restrict__ xb,   // (N_VOX+1, 32) bf16
                      const int*    __restrict__ nidx, // (N_VOX, 27)
                      const __bf16* __restrict__ wt,   // (27, COUT, 32) bf16
                      __bf16*       __restrict__ y,    // (N_VOX, COUT) bf16
                      float*        __restrict__ part) // 2*64*NSLOT floats
{
  __shared__ int s_nbr[32 * 27];

  const int tid  = threadIdx.x;
  const int lane = tid & 31;
  const int wid  = tid >> 5;
  const int ntb  = wid * 2;          // this wave's N-tile pair
  const int vb   = blockIdx.x * 32;

  // rulebook indices for 32 voxels are contiguous: coalesced stage into LDS
  for (int t = tid; t < 32 * 27; t += blockDim.x)
    s_nbr[t] = nidx[(size_t)vb * 27 + t];
  __syncthreads();

  const int m  = lane & 15;   // A row within M-tile == B column within N-tile
  const int hi = lane >> 4;   // lane half selects K chunk (ISA 16-bit A layout)
  const int k0 = hi * 8;      // lo lanes: K{0..7,16..23}; hi: K{8..15,24..31}
  const __bf16* b0base = wt + (size_t)(ntb * 16 + m) * 32;
  const __bf16* b1base = wt + (size_t)((ntb + 1) * 16 + m) * 32;

  v8f acc00 = {0.f,0.f,0.f,0.f,0.f,0.f,0.f,0.f};  // (mt=0, nt=ntb)
  v8f acc01 = acc00;                               // (mt=0, nt=ntb+1)
  v8f acc10 = acc00;                               // (mt=1, nt=ntb)
  v8f acc11 = acc00;                               // (mt=1, nt=ntb+1)

#pragma unroll
  for (int k = 0; k < 27; ++k) {
    const int r0 = s_nbr[m * 27 + k];          // mt=0 gather (N_VOX => pad row)
    const int r1 = s_nbr[(16 + m) * 27 + k];   // mt=1 gather
    const __bf16* a0row = xb + (size_t)r0 * 32;
    const __bf16* a1row = xb + (size_t)r1 * 32;
    const __bf16* b0row = b0base + (size_t)k * (COUT * 32);
    const __bf16* b1row = b1base + (size_t)k * (COUT * 32);
    ABReg a0, a1, b0, b1;
    a0.q[0] = *(const uint4*)(a0row + k0);
    a0.q[1] = *(const uint4*)(a0row + k0 + 16);
    a1.q[0] = *(const uint4*)(a1row + k0);
    a1.q[1] = *(const uint4*)(a1row + k0 + 16);
    b0.q[0] = *(const uint4*)(b0row + k0);
    b0.q[1] = *(const uint4*)(b0row + k0 + 16);
    b1.q[0] = *(const uint4*)(b1row + k0);
    b1.q[1] = *(const uint4*)(b1row + k0 + 16);
    acc00 = wmma_bf16(a0, b0, acc00);
    acc01 = wmma_bf16(a0, b1, acc01);
    acc10 = wmma_bf16(a1, b0, acc10);
    acc11 = wmma_bf16(a1, b1, acc11);
  }

  // Epilogue: store pre-BN y (bf16) + per-channel sum / sumsq partials.
  // D layout: lane half hi owns rows hi*8+r in VGPR r; channel = nt*16 + m.
  float s0 = 0.f, q0 = 0.f, s1 = 0.f, q1 = 0.f;
#pragma unroll
  for (int r = 0; r < 8; ++r) {
    const float v00 = acc00[r], v01 = acc01[r];
    const float v10 = acc10[r], v11 = acc11[r];
    s0 += v00 + v10;  q0 += v00 * v00 + v10 * v10;
    s1 += v01 + v11;  q1 += v01 * v01 + v11 * v11;
    const size_t row0 = (size_t)(vb + hi * 8 + r) * COUT;       // mt=0
    const size_t row1 = (size_t)(vb + 16 + hi * 8 + r) * COUT;  // mt=1
    y[row0 + ntb * 16 + m]       = (__bf16)v00;
    y[row0 + (ntb + 1) * 16 + m] = (__bf16)v01;
    y[row1 + ntb * 16 + m]       = (__bf16)v10;
    y[row1 + (ntb + 1) * 16 + m] = (__bf16)v11;
  }
  // combine the two row-halves of each channel (partner lane ^ 16)
  s0 += __shfl_xor(s0, 16, 32);  q0 += __shfl_xor(q0, 16, 32);
  s1 += __shfl_xor(s1, 16, 32);  q1 += __shfl_xor(q1, 16, 32);
  if (hi == 0) {
    const int slot = blockIdx.x & (NSLOT - 1);
    const int ch0  = ntb * 16 + m;
    const int ch1  = (ntb + 1) * 16 + m;
    atomicAdd(&part[ch0 * NSLOT + slot],        s0);
    atomicAdd(&part[(64 + ch0) * NSLOT + slot], q0);
    atomicAdd(&part[ch1 * NSLOT + slot],        s1);
    atomicAdd(&part[(64 + ch1) * NSLOT + slot], q1);
  }
}

// ---------------------------------------------------------------------------
// Finalize BN stats: reduce NSLOT partials per channel -> (scale, shift)
// ---------------------------------------------------------------------------
__global__ void stats_finalize_kernel(const float* __restrict__ part,
                                      const float* __restrict__ g,
                                      const float* __restrict__ b,
                                      float2* __restrict__ ss, float inv_n) {
  __shared__ float rs[NSLOT];
  __shared__ float rq[NSLOT];
  const int ch = blockIdx.x;
  const int t  = threadIdx.x;
  rs[t] = part[ch * NSLOT + t];
  rq[t] = part[(64 + ch) * NSLOT + t];
  __syncthreads();
  for (int o = NSLOT / 2; o > 0; o >>= 1) {
    if (t < o) { rs[t] += rs[t + o]; rq[t] += rq[t + o]; }
    __syncthreads();
  }
  if (t == 0) {
    const float mean = rs[0] * inv_n;
    const float var  = rq[0] * inv_n - mean * mean;
    const float rstd = rsqrtf(var + BN_EPS);
    const float sc   = rstd * g[ch];
    ss[ch] = make_float2(sc, b[ch] - mean * sc);
  }
}

// ---------------------------------------------------------------------------
// BN + ReLU (layers 1-3): y bf16 -> next-layer activations bf16, 8 elems/thread
// ---------------------------------------------------------------------------
template <int COUT>
__global__ void bn_relu_kernel(const __bf16* __restrict__ y,
                               const float2* __restrict__ ss,
                               __bf16* __restrict__ xo) {
  const int n8 = N_VOX * COUT / 8;
  const uint4* yin  = (const uint4*)y;
  uint4*       xout = (uint4*)xo;
  for (int i = blockIdx.x * blockDim.x + threadIdx.x; i < n8;
       i += gridDim.x * blockDim.x) {
    BF8 in, out;
    in.q = yin[i];
    const int chb = (i * 8) & (COUT - 1);
#pragma unroll
    for (int j = 0; j < 8; ++j) {
      const float2 p = ss[chb + j];
      out.h[j] = (__bf16)fmaxf((float)in.h[j] * p.x + p.y, 0.0f);
    }
    xout[i] = out.q;
  }
}

// ---------------------------------------------------------------------------
// Layer 4: BN + ReLU + segment max pool (ordered-int atomicMax, vals >= 0)
// ---------------------------------------------------------------------------
__global__ void bn_relu_pool_kernel(const __bf16* __restrict__ y,
                                    const float2* __restrict__ ss,
                                    const int* __restrict__ pool_ids,
                                    float* __restrict__ out) {
  const int n8 = N_VOX * C_OUT4 / 8;
  const uint4* yin = (const uint4*)y;
  for (int i = blockIdx.x * blockDim.x + threadIdx.x; i < n8;
       i += gridDim.x * blockDim.x) {
    BF8 in;
    in.q = yin[i];
    const int chb = (i * 8) & (C_OUT4 - 1);
    const int sg  = pool_ids[(i * 8) >> 6];
    int* obase = (int*)(out + (size_t)sg * C_OUT4 + chb);
#pragma unroll
    for (int j = 0; j < 8; ++j) {
      const float2 p = ss[chb + j];
      const float  v = fmaxf((float)in.h[j] * p.x + p.y, 0.0f);
      atomicMax(obase + j, __float_as_int(v));
    }
  }
}

// ---------------------------------------------------------------------------
// Launch
// ---------------------------------------------------------------------------
static inline size_t align256(size_t x) { return (x + 255) & ~(size_t)255; }

extern "C" void kernel_launch(void* const* d_in, const int* in_sizes, int n_in,
                              void* d_out, int out_size, void* d_ws,
                              size_t ws_size, hipStream_t stream) {
  (void)in_sizes; (void)n_in; (void)ws_size;

  const float* feats    = (const float*)d_in[0];
  const int*   nidx     = (const int*)d_in[1];
  const int*   pool_ids = (const int*)d_in[2];
  // d_in[3] = n_out scalar (implied by out_size); unused on device
  const float* W[4] = {(const float*)d_in[4],  (const float*)d_in[7],
                       (const float*)d_in[10], (const float*)d_in[13]};
  const float* G[4] = {(const float*)d_in[5],  (const float*)d_in[8],
                       (const float*)d_in[11], (const float*)d_in[14]};
  const float* B[4] = {(const float*)d_in[6],  (const float*)d_in[9],
                       (const float*)d_in[12], (const float*)d_in[15]};

  char* ws = (char*)d_ws;
  size_t off = 0;
  const size_t SZX = align256((size_t)(N_VOX + 1) * C_IN * sizeof(__bf16));
  __bf16* xA = (__bf16*)(ws + off); off += SZX;
  __bf16* xB = (__bf16*)(ws + off); off += SZX;
  __bf16* yb = (__bf16*)(ws + off); off += align256((size_t)N_VOX * C_OUT4 * sizeof(__bf16));
  __bf16* wt1 = (__bf16*)(ws + off); off += align256(27 * 32 * 32 * sizeof(__bf16));
  __bf16* wt2 = (__bf16*)(ws + off); off += align256(27 * 32 * 32 * sizeof(__bf16));
  __bf16* wt3 = (__bf16*)(ws + off); off += align256(27 * 32 * 32 * sizeof(__bf16));
  __bf16* wt4 = (__bf16*)(ws + off); off += align256(27 * 64 * 32 * sizeof(__bf16));
  float*  part = (float*)(ws + off);  off += align256(2 * 64 * NSLOT * sizeof(float));
  float2* ss   = (float2*)(ws + off); off += align256(64 * sizeof(float2));

  const float inv_n = 1.0f / (float)N_VOX;
  const int   nConvBlk = N_VOX / 32;          // 15625, exact

  // ---- init: output, padded activations, weights --------------------------
  zero_f32_kernel<<<4096, 256, 0, stream>>>((float*)d_out, (long)out_size);
  cast_feats_kernel<<<4096, 256, 0, stream>>>(feats, xA);
  // zero the pad row of xB (32 bf16 == 16 floats)
  zero_f32_kernel<<<1, 64, 0, stream>>>((float*)(xB + (size_t)N_VOX * C_IN), 16);
  prep_wt_kernel<<<64, 256, 0, stream>>>(W[0], wt1, 32);
  prep_wt_kernel<<<64, 256, 0, stream>>>(W[1], wt2, 32);
  prep_wt_kernel<<<64, 256, 0, stream>>>(W[2], wt3, 32);
  prep_wt_kernel<<<64, 256, 0, stream>>>(W[3], wt4, 64);

  // ---- layer 1: xA -> yb -> xB --------------------------------------------
  zero_f32_kernel<<<128, 256, 0, stream>>>(part, 2 * 64 * NSLOT);
  conv_wmma_kernel<32><<<nConvBlk, 32, 0, stream>>>(xA, nidx, wt1, yb, part);
  stats_finalize_kernel<<<32, NSLOT, 0, stream>>>(part, G[0], B[0], ss, inv_n);
  bn_relu_kernel<32><<<4096, 256, 0, stream>>>(yb, ss, xB);

  // ---- layer 2: xB -> yb -> xA --------------------------------------------
  zero_f32_kernel<<<128, 256, 0, stream>>>(part, 2 * 64 * NSLOT);
  conv_wmma_kernel<32><<<nConvBlk, 32, 0, stream>>>(xB, nidx, wt2, yb, part);
  stats_finalize_kernel<<<32, NSLOT, 0, stream>>>(part, G[1], B[1], ss, inv_n);
  bn_relu_kernel<32><<<4096, 256, 0, stream>>>(yb, ss, xA);

  // ---- layer 3: xA -> yb -> xB --------------------------------------------
  zero_f32_kernel<<<128, 256, 0, stream>>>(part, 2 * 64 * NSLOT);
  conv_wmma_kernel<32><<<nConvBlk, 32, 0, stream>>>(xA, nidx, wt3, yb, part);
  stats_finalize_kernel<<<32, NSLOT, 0, stream>>>(part, G[2], B[2], ss, inv_n);
  bn_relu_kernel<32><<<4096, 256, 0, stream>>>(yb, ss, xB);

  // ---- layer 4: xB -> yb -> BN+ReLU+segment-max pool -> d_out -------------
  zero_f32_kernel<<<128, 256, 0, stream>>>(part, 2 * 64 * NSLOT);
  conv_wmma_kernel<64><<<nConvBlk, 64, 0, stream>>>(xB, nidx, wt4, yb, part);
  stats_finalize_kernel<<<64, NSLOT, 0, stream>>>(part, G[3], B[3], ss, inv_n);
  bn_relu_pool_kernel<<<4096, 256, 0, stream>>>(yb, ss, pool_ids, (float*)d_out);
}
